// CME_84653805404525
// MI455X (gfx1250) — matrix operand order repaired and checked by reading, not compiled, not run on hardware
//
#include <hip/hip_runtime.h>
#include <hip/hip_bf16.h>
#include <cstdint>

typedef __attribute__((ext_vector_type(16))) _Float16 v16h;
typedef __attribute__((ext_vector_type(8)))  float    v8f;

#define C_IN   256
#define C8     32
#define HW     784
#define NT     128
#define EPSV   1e-5f

#if __has_builtin(__builtin_amdgcn_global_load_async_to_lds_b128)
#define HAVE_ASYNC_LDS 1
#endif

// Builtin parameter types (confirmed by toolchain diagnostics):
//   param0: int __attribute__((vector_size(16))) * in AS1 (__device__)
//   param1: same pointee in AS3 (LDS)
typedef int v4i_vs __attribute__((vector_size(4 * sizeof(int))));
typedef __attribute__((address_space(1))) v4i_vs* g128p;
typedef __attribute__((address_space(3))) v4i_vs* l128p;

// Issue one 16-byte global->LDS transfer per lane (async on gfx1250).
__device__ __forceinline__ void async_copy_b128(const void* g, void* l) {
#if defined(HAVE_ASYNC_LDS)
    __builtin_amdgcn_global_load_async_to_lds_b128(
        (g128p)(uintptr_t)g, (l128p)(unsigned)(uintptr_t)l, 0, 0);
#else
    *(float4*)l = *(const float4*)g;           // sync fallback via VGPRs
#endif
}

// Wait for staged data: ASYNCcnt <= N on the async path, DScnt==0 on fallback.
template <int N>
__device__ __forceinline__ void wait_stage() {
#if defined(HAVE_ASYNC_LDS)
#if __has_builtin(__builtin_amdgcn_s_wait_asynccnt)
    __builtin_amdgcn_s_wait_asynccnt(N);
#else
    asm volatile("s_wait_asynccnt %0" :: "i"(N) : "memory");
#endif
#else
    asm volatile("s_wait_dscnt 0" ::: "memory");
#endif
}

// Stage one 16-pixel x 32-channel fp32 slab (2KB): 4 x (32 lanes x b128).
__device__ __forceinline__ void stage_issue(const float* __restrict__ xn,
                                            int pix0, int kc, float* buf, int lane) {
    const int seg = lane & 3;        // 4 pixels (16B) per lane
    const int chb = lane >> 2;       // 8 channels per instruction
    #pragma unroll
    for (int t = 0; t < 4; ++t) {
        const int ch = t * 8 + chb;  // local channel [0,32)
        const float* g = xn + (size_t)(kc * 32 + ch) * HW + pix0 + seg * 4;
        float*       l = buf + ch * 16 + seg * 4;
        async_copy_b128(g, l);
    }
}

// ---------------------------------------------------------------------------
// Kernel 1: fused conv1x1(q,k) + BN + ReLU + global avg pool.
// Grid: 128 blocks (one per image), 256 threads (8 waves).
// Per wave: async double-buffered 16x32 fp32 x-slabs in LDS, f16 WMMA
// 16x16x32 over K=256 (8 chunks) x 4 N-tiles (q0,q1,k0,k1); fused BN/ReLU
// epilogue reduces pixel rows and accumulates pooled sums via global atomics.
// ---------------------------------------------------------------------------
__global__ __launch_bounds__(256) void k_proj_pool(
    const float* __restrict__ x,
    const float* __restrict__ wq, const float* __restrict__ bq,
    const float* __restrict__ gq, const float* __restrict__ betaq,
    const float* __restrict__ mq, const float* __restrict__ vq,
    const float* __restrict__ wk, const float* __restrict__ bk,
    const float* __restrict__ gk, const float* __restrict__ betak,
    const float* __restrict__ mk, const float* __restrict__ vk,
    float* __restrict__ qout, float* __restrict__ kout)
{
    __shared__ __attribute__((aligned(32))) _Float16 wlds[64 * 256];      // 32KB
    __shared__ __attribute__((aligned(16))) float    xbuf[8][2][32 * 16]; // 32KB

    const int tid  = threadIdx.x;
    const int wave = tid >> 5;
    const int lane = tid & 31;
    const int n    = blockIdx.x;
    const float* __restrict__ xn = x + (size_t)n * C_IN * HW;

    // cooperative weight load: rows 0-31 = wq, 32-63 = wk, converted to f16
    for (int idx = tid * 64, e = tid * 64 + 64; idx < e; ++idx) {
        int row = idx >> 8, colw = idx & 255;
        float wv = (row < 32) ? wq[row * 256 + colw] : wk[(row - 32) * 256 + colw];
        wlds[idx] = (_Float16)wv;
    }
    __syncthreads();

    const int col = lane & 15;     // N column (and A row) for this lane
    const int hi  = lane >> 4;     // half-wave index
    const int kbA = hi * 8;        // A fragment K base
    const int kbB = hi * 16;       // unused; B fragment also uses kbA

    // ---- hoist BN/bias fold: per lane, per N-tile: z = s*dot + t2 ----
    float sArr[4], tArr[4];
    #pragma unroll
    for (int t4 = 0; t4 < 4; ++t4) {
        const int o = (t4 & 1) * 16 + col;
        float g_, be_, m_, v_, bi_;
        if (t4 < 2) { g_ = gq[o]; be_ = betaq[o]; m_ = mq[o]; v_ = vq[o]; bi_ = bq[o]; }
        else        { g_ = gk[o]; be_ = betak[o]; m_ = mk[o]; v_ = vk[o]; bi_ = bk[o]; }
        const float s = g_ * rsqrtf(v_ + EPSV);
        sArr[t4] = s;
        tArr[t4] = s * bi_ + be_ - m_ * s;
    }

    float* buf0 = &xbuf[wave][0][0];
    float* buf1 = &xbuf[wave][1][0];

    for (int mt = wave; mt < 49; mt += 8) {
        v8f acc[4] = {};
        const int pix0 = mt * 16;

        stage_issue(xn, pix0, 0, buf0, lane);

        for (int kc = 0; kc < 8; ++kc) {        // 8 K-chunks of 32 channels
            float* cur = (kc & 1) ? buf1 : buf0;
            float* nxt = (kc & 1) ? buf0 : buf1;

            // prior chunk's LDS reads must land before overwriting its buffer
            asm volatile("s_wait_dscnt 0" ::: "memory");
            if (kc < 7) {
                stage_issue(xn, pix0, kc + 1, nxt, lane);
                wait_stage<4>();                 // chunk kc done, kc+1 in flight
            } else {
                wait_stage<0>();
            }

            // A fragment: A[m][k] = cur[k*16 + m], f32 -> f16
            v16h a;
            #pragma unroll
            for (int j = 0; j < 8; ++j) {
                a[j]     = (_Float16)cur[(kbA + j) * 16 + col];
                a[8 + j] = (_Float16)cur[(16 + kbA + j) * 16 + col];
            }

            const int kg = kc * 32;              // global K offset
            #pragma unroll
            for (int t4 = 0; t4 < 4; ++t4) {
                v16h bfrag;
                const _Float16* bp = wlds + (t4 * 16 + col) * 256 + kg + kbA;
                #pragma unroll
                for (int j = 0; j < 8; ++j) { bfrag[j] = bp[j]; bfrag[8 + j] = bp[16 + j]; }
                acc[t4] = __builtin_amdgcn_wmma_f32_16x16x32_f16(
                    false, a, false, bfrag, (short)0, acc[t4], false, false);
            }
        }

        // ---- epilogue: affine + ReLU, reduce the 16 pixel rows, atomics ----
        #pragma unroll
        for (int t4 = 0; t4 < 4; ++t4) {
            float part = 0.0f;
            #pragma unroll
            for (int r = 0; r < 8; ++r)
                part += fmaxf(acc[t4][r] * sArr[t4] + tArr[t4], 0.0f);
            part += __shfl_down(part, 16, 32);   // fold lanes 16-31 (M=8..15)
            if (lane < 16) {
                const int o = (t4 & 1) * 16 + col;
                float* dst = (t4 < 2 ? qout : kout) + n * C8 + o;
                unsafeAtomicAdd(dst, part * (1.0f / 784.0f));
            }
        }
    }
}

// ---------------------------------------------------------------------------
// Kernel 2: attention (softmax over axis=1) + q_upd + inflate GEMM (WMMA)
// + BN + ReLU + sigmoid -> gate (128 x 256). One block, 256 threads.
// ---------------------------------------------------------------------------
__global__ __launch_bounds__(256) void k_att_gate(
    const float* __restrict__ qbuf, const float* __restrict__ kbuf,
    const float* __restrict__ wi, const float* __restrict__ bi,
    const float* __restrict__ gi, const float* __restrict__ betai,
    const float* __restrict__ mi, const float* __restrict__ vi,
    float* __restrict__ gate)
{
    __shared__ float    qs[NT * C8];
    __shared__ float    ks[NT * C8];
    __shared__ float    att[16 * 8 * 8];
    __shared__ __attribute__((aligned(32))) _Float16 qup[NT * C8];

    const int tid = threadIdx.x;

    for (int i = tid; i < NT * C8; i += 256) { qs[i] = qbuf[i]; ks[i] = kbuf[i]; }
    __syncthreads();

    // att[b,i,j] = -<q[b,i,:], q[b,j,:]>
    for (int idx = tid; idx < 1024; idx += 256) {
        int b = idx >> 6, i = (idx >> 3) & 7, j = idx & 7;
        const float* qi = qs + (b * 8 + i) * C8;
        const float* qj = qs + (b * 8 + j) * C8;
        float d = 0.0f;
        #pragma unroll
        for (int c = 0; c < C8; ++c) d += qi[c] * qj[c];
        att[idx] = -d;
    }
    __syncthreads();

    // softmax over axis=1 (the i index): one thread per (b,j)
    if (tid < 128) {
        int b = tid >> 3, j = tid & 7;
        float mx = -1e30f;
        #pragma unroll
        for (int i = 0; i < 8; ++i) mx = fmaxf(mx, att[b * 64 + i * 8 + j]);
        float e[8], sm = 0.0f;
        #pragma unroll
        for (int i = 0; i < 8; ++i) { e[i] = __expf(att[b * 64 + i * 8 + j] - mx); sm += e[i]; }
        const float inv = 1.0f / sm;
        #pragma unroll
        for (int i = 0; i < 8; ++i) att[b * 64 + i * 8 + j] = e[i] * inv;
    }
    __syncthreads();

    // q_upd[b,i,c] = sum_j att[b,i,j] * v[b,j,c] + v[b,i,c]  (v = k branch)
    for (int idx = tid; idx < NT * C8; idx += 256) {
        int row = idx >> 5, c = idx & 31;
        int b = row >> 3, i = row & 7;
        float sum = ks[row * C8 + c];
        #pragma unroll
        for (int j = 0; j < 8; ++j)
            sum += att[b * 64 + i * 8 + j] * ks[(b * 8 + j) * C8 + c];
        qup[idx] = (_Float16)sum;
    }
    __syncthreads();

    // inflate: y = q_upd @ wi.T + bi ; BN ; ReLU ; sigmoid -> gate
    const int wave = tid >> 5, lane = tid & 31;
    const int col = lane & 15, hi = lane >> 4;
    const int kbA = hi * 8, kbB = hi * 16;
    const int mt = wave;                       // 8 waves x 16 rows = 128 rows

    v16h a;
    const _Float16* ap = qup + (mt * 16 + col) * C8 + kbA;
    #pragma unroll
    for (int j = 0; j < 8; ++j) { a[j] = ap[j]; a[8 + j] = ap[16 + j]; }

    for (int nt = 0; nt < 16; ++nt) {
        const int o = nt * 16 + col;
        v16h bfrag;
        const float* bp = wi + o * C8 + kbA;   // wi is (256, 32) row-major
        #pragma unroll
        for (int j = 0; j < 8; ++j) { bfrag[j] = (_Float16)bp[j]; bfrag[8 + j] = (_Float16)bp[16 + j]; }
        v8f acc = {};
        acc = __builtin_amdgcn_wmma_f32_16x16x32_f16(
            false, a, false, bfrag, (short)0, acc, false, false);

        const float s  = gi[o] * rsqrtf(vi[o] + EPSV);
        const float t2 = s * bi[o] + betai[o] - mi[o] * s;
        #pragma unroll
        for (int r = 0; r < 8; ++r) {
            const int row = mt * 16 + r + hi * 8;
            float z = fmaxf(acc[r] * s + t2, 0.0f);
            gate[row * C_IN + o] = 1.0f / (1.0f + __expf(-z));
        }
    }
}

// ---------------------------------------------------------------------------
// Kernel 3: out[n,c,h,w] = gate[n,c] * x[n,c,h,w]  (streaming, float4)
// ---------------------------------------------------------------------------
__global__ __launch_bounds__(256) void k_gate_mul(
    const float* __restrict__ x, const float* __restrict__ gate,
    float* __restrict__ out)
{
    const int i   = blockIdx.x * 256 + threadIdx.x;  // float4 index
    const int row = i / 196;                          // (n*256 + c)
    const float g = gate[row];
    const float4 xv = ((const float4*)x)[i];
    float4 ov;
    ov.x = xv.x * g; ov.y = xv.y * g; ov.z = xv.z * g; ov.w = xv.w * g;
    ((float4*)out)[i] = ov;
}

// ---------------------------------------------------------------------------
extern "C" void kernel_launch(void* const* d_in, const int* in_sizes, int n_in,
                              void* d_out, int out_size, void* d_ws, size_t ws_size,
                              hipStream_t stream) {
    (void)in_sizes; (void)n_in; (void)out_size; (void)ws_size;
    const float* x     = (const float*)d_in[0];
    const float* wq    = (const float*)d_in[1];
    const float* bq    = (const float*)d_in[2];
    const float* gq    = (const float*)d_in[3];
    const float* betaq = (const float*)d_in[4];
    const float* mq    = (const float*)d_in[5];
    const float* vq    = (const float*)d_in[6];
    const float* wk    = (const float*)d_in[7];
    const float* bk    = (const float*)d_in[8];
    const float* gk    = (const float*)d_in[9];
    const float* betak = (const float*)d_in[10];
    const float* mk    = (const float*)d_in[11];
    const float* vk    = (const float*)d_in[12];
    const float* wi    = (const float*)d_in[13];
    const float* bi    = (const float*)d_in[14];
    const float* gi    = (const float*)d_in[15];
    const float* betai = (const float*)d_in[16];
    const float* mi    = (const float*)d_in[17];
    const float* vi    = (const float*)d_in[18];

    float* ws   = (float*)d_ws;
    float* qbuf = ws;                 // 128*32, atomically accumulated
    float* kbuf = ws + NT * C8;       // 128*32
    float* gate = ws + 2 * NT * C8;   // 128*256

    (void)hipMemsetAsync(d_ws, 0, (size_t)2 * NT * C8 * sizeof(float), stream);

    k_proj_pool<<<NT, 256, 0, stream>>>(x, wq, bq, gq, betaq, mq, vq,
                                        wk, bk, gk, betak, mk, vk, qbuf, kbuf);
    k_att_gate<<<1, 256, 0, stream>>>(qbuf, kbuf, wi, bi, gi, betai, mi, vi, gate);
    k_gate_mul<<<(NT * C_IN * 196) / 256, 256, 0, stream>>>(x, gate, (float*)d_out);
}
